// VectorQuantizer_78434692759724
// MI455X (gfx1250) — matrix operand order repaired and checked by reading, not compiled, not run on hardware
//
#include <hip/hip_runtime.h>
#include <hip/hip_bf16.h>
#include <math.h>

typedef __attribute__((ext_vector_type(16))) __bf16 v16bf;
typedef __attribute__((ext_vector_type(8)))  float  v8f;

#define XS_STRIDE 132   // 64-bank LDS: 16*132 % 64 == 0 -> A-fragment halves (16 channels apart) hit the same banks

// ---- gfx1250 async global->LDS path (ASYNCcnt), with scalar fallback ----
#if defined(__has_builtin)
# if __has_builtin(__builtin_amdgcn_global_load_async_to_lds_b128)
#  define HAVE_ASYNC_LDS 1
# endif
#endif
#ifndef HAVE_ASYNC_LDS
# define HAVE_ASYNC_LDS 0
#endif

#if HAVE_ASYNC_LDS
typedef __attribute__((__vector_size__(16))) int v4i;
typedef __attribute__((address_space(1))) v4i* gv4p;   // global (prints as "__device__")
typedef __attribute__((address_space(3))) v4i* lv4p;   // LDS    (prints as "__shared__")
__device__ __forceinline__ void async_g2l_16B(const float* g, float* l) {
  __builtin_amdgcn_global_load_async_to_lds_b128((gv4p)g, (lv4p)l, 0, 0);
}
__device__ __forceinline__ void wait_async0() {
# if __has_builtin(__builtin_amdgcn_s_wait_asynccnt)
  __builtin_amdgcn_s_wait_asynccnt(0);
# else
  asm volatile("s_wait_asynccnt 0x0" ::: "memory");
# endif
}
#endif

__device__ __forceinline__ unsigned short bf16_rne(float f) {
  union { float f; unsigned u; } x; x.f = f;
  unsigned u = x.u;
  unsigned r = u + 0x7FFFu + ((u >> 16) & 1u);
  return (unsigned short)(r >> 16);
}
__device__ __forceinline__ float bf16_to_f32(unsigned short h) {
  union { unsigned u; float f; } x; x.u = ((unsigned)h) << 16; return x.f;
}

union U16 { v16bf v; unsigned short s[16]; };

// ---------- Kernel 0: per-code norms, bf16 hi/lo split of E, zero accumulators ----------
__global__ void vq_prep(const float* __restrict__ emb, float* __restrict__ lossAcc,
                        unsigned* __restrict__ hist, float* __restrict__ enorm,
                        unsigned short* __restrict__ Ehi, unsigned short* __restrict__ Elo) {
  int k = threadIdx.x;                       // 512 threads
  if (k == 0) lossAcc[0] = 0.0f;
  if (k < 512) {
    hist[k] = 0u;
    float s = 0.0f;
    for (int c = 0; c < 64; ++c) {
      float e = emb[k * 64 + c];
      s += e * e;
      unsigned short hb = bf16_rne(e);
      Ehi[k * 64 + c] = hb;
      Elo[k * 64 + c] = bf16_rne(e - bf16_to_f32(hb));
    }
    enorm[k] = s;
  }
}

// ---------- Kernel 1: WMMA distance GEMM + argmin + quantize + loss + histogram ----------
// 128 threads = 4 waves; each wave owns 2 slabs of 16 tokens (32 tokens/wave, 128/block).
__global__ __launch_bounds__(128)
void vq_main(const float* __restrict__ x, const float* __restrict__ emb,
             const float* __restrict__ enorm,
             const unsigned short* __restrict__ Ehi, const unsigned short* __restrict__ Elo,
             float* __restrict__ out, float* __restrict__ lossAcc, unsigned* __restrict__ hist) {
  __shared__ float xs[64 * XS_STRIDE];   // X tile, channel-major [c][token]
  __shared__ float enormS[512];
  __shared__ int   sidx[128];
  __shared__ float red[128];

  const int tid = threadIdx.x;
  const int blk = blockIdx.x;                    // 2048 blocks, 128 tokens each
  const int b       = blk >> 8;                  // 256 blocks per batch element
  const int thwBase = (blk & 255) << 7;
  const long baseIO = (long)b * (64L * 32768L) + thwBase;

  // Stage A: stage the 64x128 token tile into LDS
#if HAVE_ASYNC_LDS
  for (int i = tid; i < 64 * 32; i += 128) {     // 16B chunks: 4 tokens each
    int c = i >> 5, tg = (i & 31) << 2;
    async_g2l_16B(x + baseIO + (long)c * 32768 + tg, &xs[c * XS_STRIDE + tg]);
  }
#else
  for (int i = tid; i < 64 * 128; i += 128) {
    int c = i >> 7, t = i & 127;
    xs[c * XS_STRIDE + t] = x[baseIO + (long)c * 32768 + t];
  }
#endif
  for (int j = tid; j < 512; j += 128) enormS[j] = enorm[j];
#if HAVE_ASYNC_LDS
  wait_async0();
#endif
  __syncthreads();

  const int lane   = tid & 31;
  const int wave   = tid >> 5;
  const int halfId = lane >> 4;
  const int col    = lane & 15;

  // A fragments for both slabs (contiguous k = 16*half + s, matching B), bf16 hi/lo split
  U16 ah[2][2], al[2][2];                        // [slab][k-chunk]
  #pragma unroll
  for (int s = 0; s < 2; ++s) {
    const int tokA = (wave << 5) + (s << 4) + col;   // A row M = lane%16
    #pragma unroll
    for (int q = 0; q < 2; ++q) {
      #pragma unroll
      for (int v = 0; v < 8; ++v) {
        int kp = 2 * v + 16 * halfId;
        int c0 = q * 32 + kp;
        float f0 = xs[c0 * XS_STRIDE + tokA];
        float f1 = xs[(c0 + 1) * XS_STRIDE + tokA];
        unsigned short h0 = bf16_rne(f0), h1 = bf16_rne(f1);
        ah[s][q].s[2 * v]     = h0;
        ah[s][q].s[2 * v + 1] = h1;
        al[s][q].s[2 * v]     = bf16_rne(f0 - bf16_to_f32(h0));
        al[s][q].s[2 * v + 1] = bf16_rne(f1 - bf16_to_f32(h1));
      }
    }
  }

  float bestD[2][8];
  int   bestI[2][8];
  #pragma unroll
  for (int s = 0; s < 2; ++s)
    #pragma unroll
    for (int r = 0; r < 8; ++r) { bestD[s][r] = 3.4e38f; bestI[s][r] = 0; }

  // Sweep 32 code tiles; one B-fragment load feeds 12 WMMAs (2 slabs x 6)
  #pragma unroll 2
  for (int ct = 0; ct < 32; ++ct) {
    const int code = (ct << 4) + col;                 // B column N = lane%16
    const unsigned short* eh = Ehi + code * 64 + 16 * halfId;
    const unsigned short* el = Elo + code * 64 + 16 * halfId;
    const v16bf bh0 = *(const v16bf*)(eh);
    const v16bf bl0 = *(const v16bf*)(el);
    const v16bf bh1 = *(const v16bf*)(eh + 32);
    const v16bf bl1 = *(const v16bf*)(el + 32);
    const float en  = enormS[code];

    #pragma unroll
    for (int s = 0; s < 2; ++s) {
      v8f acc = {};
      acc = __builtin_amdgcn_wmma_f32_16x16x32_bf16(false, ah[s][0].v, false, bh0, (short)0, acc, false, false);
      acc = __builtin_amdgcn_wmma_f32_16x16x32_bf16(false, ah[s][0].v, false, bl0, (short)0, acc, false, false);
      acc = __builtin_amdgcn_wmma_f32_16x16x32_bf16(false, al[s][0].v, false, bh0, (short)0, acc, false, false);
      acc = __builtin_amdgcn_wmma_f32_16x16x32_bf16(false, ah[s][1].v, false, bh1, (short)0, acc, false, false);
      acc = __builtin_amdgcn_wmma_f32_16x16x32_bf16(false, ah[s][1].v, false, bl1, (short)0, acc, false, false);
      acc = __builtin_amdgcn_wmma_f32_16x16x32_bf16(false, al[s][1].v, false, bh1, (short)0, acc, false, false);

      // codes ascend per lane -> strict '<' == first-occurrence argmin (no tie-break needed here)
      #pragma unroll
      for (int r = 0; r < 8; ++r) {
        float d = en - 2.0f * acc[r];                 // ||x||^2 dropped: per-token constant
        if (d < bestD[s][r]) { bestD[s][r] = d; bestI[s][r] = code; }
      }
    }
  }

  // Cross-lane argmin (codes live across the 16 lanes of each half); tie-break on index here
  #pragma unroll
  for (int m = 1; m < 16; m <<= 1) {
    #pragma unroll
    for (int s = 0; s < 2; ++s)
      #pragma unroll
      for (int r = 0; r < 8; ++r) {
        float od = __shfl_xor(bestD[s][r], m, 32);
        int   oi = __shfl_xor(bestI[s][r], m, 32);
        if (od < bestD[s][r] || (od == bestD[s][r] && oi < bestI[s][r])) {
          bestD[s][r] = od; bestI[s][r] = oi;
        }
      }
  }
  if (col == 0) {
    #pragma unroll
    for (int s = 0; s < 2; ++s)
      #pragma unroll
      for (int r = 0; r < 8; ++r)                      // D row M = r + 8*half
        sidx[(wave << 5) + (s << 4) + 8 * halfId + r] = bestI[s][r];
  }
  __syncthreads();

  atomicAdd(&hist[sidx[tid]], 1u);                     // 128 tokens, 128 threads

  // Stage C: coalesced quantized writes + exact fp32 MSE partial
  float lsum = 0.0f;
  for (int i = tid; i < 64 * 128; i += 128) {
    int c = i >> 7, t = i & 127;
    float q  = emb[sidx[t] * 64 + c];
    float xv = xs[c * XS_STRIDE + t];
    float df = q - xv;
    lsum += df * df;
    out[baseIO + (long)c * 32768 + t] = q;             // quantized_st forward value == quantized
  }
  red[tid] = lsum;
  __syncthreads();
  for (int s = 64; s > 0; s >>= 1) {
    if (tid < s) red[tid] += red[tid + s];
    __syncthreads();
  }
  if (tid == 0) atomicAdd(lossAcc, red[0]);
}

// ---------- Kernel 2: loss scale + perplexity ----------
__global__ void vq_final(const unsigned* __restrict__ hist, const float* __restrict__ lossAcc,
                         float* __restrict__ out) {
  __shared__ float red[512];
  int k = threadIdx.x;
  float p = (float)hist[k] * (1.0f / 262144.0f);
  red[k] = p * logf(p + 1e-10f);
  __syncthreads();
  for (int s = 256; s > 0; s >>= 1) {
    if (k < s) red[k] += red[k + s];
    __syncthreads();
  }
  if (k == 0) {
    out[16777216] = lossAcc[0] * (1.25f / 16777216.0f);  // q_latent + 0.25*e_latent (same fwd value)
    out[16777217] = expf(-red[0]);
  }
}

extern "C" void kernel_launch(void* const* d_in, const int* in_sizes, int n_in,
                              void* d_out, int out_size, void* d_ws, size_t ws_size,
                              hipStream_t stream) {
  const float* x   = (const float*)d_in[0];   // (8,64,32,32,32) f32
  const float* emb = (const float*)d_in[1];   // (512,64) f32
  float* out = (float*)d_out;                 // 16777216 quantized + loss + perplexity

  float*          wsf   = (float*)d_ws;
  float*          lossA = wsf;                               // [0]
  unsigned*       hist  = (unsigned*)(wsf + 8);              // 512
  float*          enorm = (float*)(hist + 512);              // 512
  unsigned short* Ehi   = (unsigned short*)(enorm + 512);    // 512*64 (32B-aligned)
  unsigned short* Elo   = Ehi + 512 * 64;

  vq_prep <<<1,    512, 0, stream>>>(emb, lossA, hist, enorm, Ehi, Elo);
  vq_main <<<2048, 128, 0, stream>>>(x, emb, enorm, Ehi, Elo, out, lossA, hist);
  vq_final<<<1,    512, 0, stream>>>(hist, lossA, out);
}
